// SelfAttention_4767413698751
// MI455X (gfx1250) — matrix operand order repaired and checked by reading, not compiled, not run on hardware
//
#include <hip/hip_runtime.h>
#include <hip/hip_bf16.h>

// ---------------------------------------------------------------------------
// Self-attention with RoPE for MI455X (gfx1250), wave32.
// - All matmuls on v_wmma_f32_16x16x32_f16 (f32 accumulate)
// - Global->LDS staging via CDNA5 async-to-LDS DMA (ASYNCcnt), double-buffered
// - Softmax f32, attn matrix written f32 (memory-bound floor ~537MB @23.3TB/s)
// ---------------------------------------------------------------------------

#define B_     2
#define N_     2048
#define D_     1024
#define H_     16
#define DH_    64
#define DR_    32
#define INNER_ 1024
#define SCALE_ 0.125f  // DH^-0.5

typedef __attribute__((ext_vector_type(16))) _Float16 v16h;
typedef __attribute__((ext_vector_type(8)))  float    v8f;

union FragBits { unsigned int u[8]; v16h h; };
union Pack4h   { _Float16 h[4]; unsigned long long u64; };

// ---- CDNA5 async/LDS primitives ----------------------------------------- //
// One 16B global->LDS DMA transfer per lane; tracked by ASYNCcnt.
__device__ __forceinline__ void async_load_b128(void* lds, const void* gsrc) {
  asm volatile("global_load_async_to_lds_b128 %0, %1, off"
               :: "v"((unsigned)(size_t)lds),
                  "v"((unsigned long long)(size_t)gsrc)
               : "memory");
}
// Wait until this wave's outstanding async ops <= CNT (loads retire in order,
// so partial waits release exactly the oldest tile).
template <int CNT>
__device__ __forceinline__ void wait_async() {
  asm volatile("s_wait_asynccnt %0" :: "n"(CNT) : "memory");
}
__device__ __forceinline__ void lds_wave_fence() {
  asm volatile("s_wait_dscnt 0" ::: "memory");
}

// ---- WMMA fragment builders (ISA §7.12.2 layouts) ------------------------ //
// A 16x32 f16: lane row = lane&15, half-group g = lane>>4; K halves land as
// two contiguous 16B runs (8g..8g+7 and 16+8g..23+8g) -> two b128 LDS loads.
__device__ __forceinline__ v16h load_frag_a16(const _Float16* base, int ld,
                                              int lane, int k0) {
  const int row = lane & 15, g = lane >> 4;
  FragBits f;
#pragma unroll
  for (int j = 0; j < 8; ++j) {
    const int k = k0 + ((j < 4) ? (2 * j + 8 * g) : (16 + 2 * (j - 4) + 8 * g));
    f.u[j] = *(const unsigned int*)(base + row * ld + k);
  }
  return f.h;
}
// B 32x16 f16 from an LDS tile stored [n][k]: same K order as A fragment
// (8g..8g+7 and 16+8g..23+8g).
__device__ __forceinline__ v16h load_frag_b16(const _Float16* base, int ld,
                                              int lane, int k0) {
  const int n = lane & 15, g = lane >> 4;
  FragBits f;
#pragma unroll
  for (int t = 0; t < 8; ++t)
    f.u[t] = *(const unsigned int*)(base + n * ld + k0 + ((t < 4) ? (2 * t + 8 * g) : (16 + 2 * (t - 4) + 8 * g)));
  return f.h;
}

// ---------------------------------------------------------------------------
// Kernel 1: f32 -> f16 convert (x4 vectorized; all sizes divisible by 4)
// ---------------------------------------------------------------------------
__global__ void k_f32_to_f16(const float* __restrict__ src,
                             _Float16* __restrict__ dst, int n4) {
  int i = blockIdx.x * blockDim.x + threadIdx.x;
  if (i >= n4) return;
  const float4 v = ((const float4*)src)[i];
  Pack4h p;
  p.h[0] = (_Float16)v.x; p.h[1] = (_Float16)v.y;
  p.h[2] = (_Float16)v.z; p.h[3] = (_Float16)v.w;
  ((unsigned long long*)dst)[i] = p.u64;
}

// ---------------------------------------------------------------------------
// Kernel 2: WMMA GEMM  C[m,n] = sum_k A[m,k] * W[n,k] (+ bias[n])
// A: [M,K] f16, W: [Nn,K] f16, C: [M,Nn] f32.  64x64 block tile, 4 waves,
// K-step 32, async-staged + double-buffered LDS.
// ---------------------------------------------------------------------------
__device__ __forceinline__ void gemm_stage(const _Float16* A, const _Float16* W,
                                           _Float16* AsB, _Float16* WsB,
                                           int m0, int n0, int K, int k0,
                                           int tid) {
  // 64 rows x 4 chunks of 8 halves each; counted loops -> no exec masking.
#pragma unroll
  for (int it = 0; it < 2; ++it) {
    const int i = tid + it * 128;
    const int r = i >> 2, c = (i & 3) * 8;
    async_load_b128(AsB + r * 32 + c, A + (size_t)(m0 + r) * K + k0 + c);
  }
#pragma unroll
  for (int it = 0; it < 2; ++it) {
    const int i = tid + it * 128;
    const int r = i >> 2, c = (i & 3) * 8;
    async_load_b128(WsB + r * 32 + c, W + (size_t)(n0 + r) * K + k0 + c);
  }
}

__global__ __launch_bounds__(128)
void k_gemm_wmma(const _Float16* __restrict__ A, const _Float16* __restrict__ W,
                 float* __restrict__ C, const float* __restrict__ bias,
                 int M, int K, int Nn) {
  __shared__ _Float16 As[2][64 * 32];
  __shared__ _Float16 Ws[2][64 * 32];

  const int tid = threadIdx.x, lane = tid & 31, wv = tid >> 5;
  const int m0 = blockIdx.y * 64, n0 = blockIdx.x * 64;
  const int nk = K / 32;

  v8f acc[4] = {};

  gemm_stage(A, W, As[0], Ws[0], m0, n0, K, 0, tid);
  for (int kt = 0; kt < nk; ++kt) {
    wait_async<0>();     // this wave's share of tile kt landed in LDS
    __syncthreads();     // everyone's share landed
    if (kt + 1 < nk)     // DMA next tile while we do matrix math
      gemm_stage(A, W, As[(kt + 1) & 1], Ws[(kt + 1) & 1], m0, n0, K,
                 (kt + 1) * 32, tid);

    const _Float16* Asb = As[kt & 1];
    const _Float16* Wsb = Ws[kt & 1];
    const v16h a = load_frag_a16(Asb + wv * 16 * 32, 32, lane, 0);
#pragma unroll
    for (int nt = 0; nt < 4; ++nt) {
      const v16h b = load_frag_b16(Wsb + nt * 16 * 32, 32, lane, 0);
      acc[nt] = __builtin_amdgcn_wmma_f32_16x16x32_f16(
          false, a, false, b, (short)0, acc[nt], false, false);
    }
    __syncthreads();     // all waves consumed tile kt before it is overwritten
  }

  // D layout: lane = n + 16*(m>=8), vgpr r -> m = r + 8*(lane>>4)
  const int nl = lane & 15, mg = lane >> 4;
#pragma unroll
  for (int nt = 0; nt < 4; ++nt) {
    const int n = n0 + nt * 16 + nl;
    const float bv = bias ? bias[n] : 0.0f;
#pragma unroll
    for (int r = 0; r < 8; ++r) {
      const int m = m0 + wv * 16 + r + 8 * mg;
      C[(size_t)m * Nn + n] = acc[nt][r] + bv;
    }
  }
}

// ---------------------------------------------------------------------------
// Kernel 3: RoPE + repack.  qf:[B*N,INNER], kvf:[B*N,2*INNER] f32.
// qh/kh f16 [B,H,N,DH];  V pre-transposed: vt f16 [B,H,DH,N] so that the
// attention PV B-operand can be async-DMA'd without an LDS transpose.
// ---------------------------------------------------------------------------
__global__ void k_rope_pack(const float* __restrict__ qf,
                            const float* __restrict__ kvf,
                            const float* __restrict__ psin,
                            const float* __restrict__ pcos,
                            _Float16* __restrict__ qh,
                            _Float16* __restrict__ kh,
                            _Float16* __restrict__ vt) {
  const int total = B_ * N_ * H_ * (DH_ / 2);
  int idx = blockIdx.x * blockDim.x + threadIdx.x;
  if (idx >= total) return;

  const int p = idx & 31;
  const int h = (idx >> 5) & 15;
  const int n = (idx >> 9) & (N_ - 1);
  const int b = idx >> 20;
  const int d0 = 2 * p, d1 = d0 + 1;

  const size_t tok = (size_t)b * N_ + n;
  const size_t qoff  = tok * INNER_ + h * DH_;
  const size_t kvoff = tok * (2 * INNER_) + h * DH_;

  float q0 = qf[qoff + d0], q1 = qf[qoff + d1];
  float kk0 = kvf[kvoff + d0], kk1 = kvf[kvoff + d1];
  const float v0 = kvf[kvoff + INNER_ + d0], v1 = kvf[kvoff + INNER_ + d1];

  if (d0 < DR_) {  // t*cos + rotate_every_two(t)*sin
    const size_t ro = tok * DR_;
    const float s0 = psin[ro + d0], s1 = psin[ro + d1];
    const float c0 = pcos[ro + d0], c1 = pcos[ro + d1];
    const float qr0 = q0 * c0 - q1 * s0, qr1 = q1 * c1 + q0 * s1;
    const float kr0 = kk0 * c0 - kk1 * s0, kr1 = kk1 * c1 + kk0 * s1;
    q0 = qr0; q1 = qr1; kk0 = kr0; kk1 = kr1;
  }

  const int bh = b * H_ + h;
  const size_t ho = ((size_t)bh * N_ + n) * DH_;
  qh[ho + d0] = (_Float16)q0;  qh[ho + d1] = (_Float16)q1;
  kh[ho + d0] = (_Float16)kk0; kh[ho + d1] = (_Float16)kk1;
  vt[((size_t)bh * DH_ + d0) * N_ + n] = (_Float16)v0;
  vt[((size_t)bh * DH_ + d1) * N_ + n] = (_Float16)v1;
}

// ---------------------------------------------------------------------------
// Kernel 4: attention. One block = one (bh, 16-row tile); 128 threads/4 waves.
// LDS: S[16][2048] f32 (128K) + Sh f16 copy of P (64K) + staging (~27K).
// ---------------------------------------------------------------------------
__global__ __launch_bounds__(128)
void k_attention(const _Float16* __restrict__ qh, const _Float16* __restrict__ kh,
                 const _Float16* __restrict__ vt, float* __restrict__ attn_out,
                 _Float16* __restrict__ ctx) {
  extern __shared__ char smem[];
  float*    S   = (float*)smem;                      // 131072 B
  _Float16* Sh  = (_Float16*)(smem + 131072);        //  65536 B (P in f16)
  _Float16* Qs  = (_Float16*)(smem + 196608);        //   2048 B
  _Float16* Ks  = (_Float16*)(smem + 198656);        //  16384 B (4 wv x 2 buf)
  _Float16* Vs  = (_Float16*)(smem + 215040);        //   8192 B (4 wv x 2 buf)
  float*    red = (float*)(smem + 223232);           //   1024 B

  const int tid = threadIdx.x, lane = tid & 31, wv = tid >> 5;
  const int bh = blockIdx.x >> 7;
  const int i0 = (blockIdx.x & 127) * 16;

  // ---- stage Q tile (16 x 64 halves = 128 x 16B chunks, one per thread) ----
  const _Float16* Qg = qh + ((size_t)bh * N_ + i0) * DH_;
  {
    const int r = tid >> 3, c = (tid & 7) * 8;
    async_load_b128(Qs + r * DH_ + c, Qg + r * DH_ + c);
  }
  wait_async<0>();
  __syncthreads();
  const v16h aq0 = load_frag_a16(Qs, DH_, lane, 0);
  const v16h aq1 = load_frag_a16(Qs, DH_, lane, 32);

  // ---- S = Q K^T * scale; per-wave column tiles, double-buffered DMA ----
  _Float16* Ksw = Ks + wv * 2048;  // halves: 2 buffers x 1024
  const _Float16* Kbase = kh + (size_t)bh * N_ * DH_;
  // issue first tile (16 rows x 64 halves = 128 chunks / 32 lanes = 4 ops)
#pragma unroll
  for (int it = 0; it < 4; ++it) {
    const int i = lane + it * 32;
    const int r = i >> 3, c = (i & 7) * 8;
    async_load_b128(Ksw + r * DH_ + c, Kbase + (size_t)(wv * 16 + r) * DH_ + c);
  }
  int pbuf = 0;
  for (int jt = wv; jt < N_ / 16; jt += 4) {
    const bool more = (jt + 4) < (N_ / 16);
    if (more) {  // DMA next tile into the other buffer while computing
      _Float16* dst = Ksw + (pbuf ^ 1) * 1024;
#pragma unroll
      for (int it = 0; it < 4; ++it) {
        const int i = lane + it * 32;
        const int r = i >> 3, c = (i & 7) * 8;
        async_load_b128(dst + r * DH_ + c,
                        Kbase + (size_t)((jt + 4) * 16 + r) * DH_ + c);
      }
      wait_async<4>();   // oldest 4 ops (current tile) retired, next in flight
    } else {
      wait_async<0>();
    }
    const _Float16* Kt = Ksw + pbuf * 1024;
    const v16h b0 = load_frag_b16(Kt, DH_, lane, 0);
    const v16h b1 = load_frag_b16(Kt, DH_, lane, 32);
    v8f acc = {};
    acc = __builtin_amdgcn_wmma_f32_16x16x32_f16(false, aq0, false, b0,
                                                 (short)0, acc, false, false);
    acc = __builtin_amdgcn_wmma_f32_16x16x32_f16(false, aq1, false, b1,
                                                 (short)0, acc, false, false);
    const int nl = lane & 15, mg = lane >> 4;
    const int j0 = jt * 16;
#pragma unroll
    for (int r = 0; r < 8; ++r)
      S[(r + 8 * mg) * N_ + j0 + nl] = acc[r] * SCALE_;
    lds_wave_fence();    // frag reads done before buffer reuse
    pbuf ^= 1;
  }
  __syncthreads();

  // ---- row softmax (f32): 8 threads/row, 256 cols each, float4 passes ----
  const int row = tid >> 3;
  const int c0 = (tid & 7) * 256;
  float* Sr = S + row * N_;
  float mx = -3.4e38f;
  for (int c = c0; c < c0 + 256; c += 4) {
    const float4 v = *(const float4*)&Sr[c];
    mx = fmaxf(mx, fmaxf(fmaxf(v.x, v.y), fmaxf(v.z, v.w)));
  }
  red[tid] = mx;
  __syncthreads();
  if ((tid & 7) == 0) {
    float m2 = red[tid];
#pragma unroll
    for (int t = 1; t < 8; ++t) m2 = fmaxf(m2, red[tid + t]);
    red[128 + row] = m2;
  }
  __syncthreads();
  mx = red[128 + row];
  float sum = 0.0f;
  for (int c = c0; c < c0 + 256; c += 4) {
    float4 v = *(const float4*)&Sr[c];
    v.x = __expf(v.x - mx); v.y = __expf(v.y - mx);
    v.z = __expf(v.z - mx); v.w = __expf(v.w - mx);
    *(float4*)&Sr[c] = v;
    sum += (v.x + v.y) + (v.z + v.w);
  }
  __syncthreads();
  red[tid] = sum;
  __syncthreads();
  if ((tid & 7) == 0) {
    float s2 = 0.0f;
#pragma unroll
    for (int t = 0; t < 8; ++t) s2 += red[tid + t];
    red[144 + row] = s2;
  }
  __syncthreads();
  const float inv = 1.0f / red[144 + row];
  float* attn_row = attn_out + ((size_t)bh * N_ + i0 + row) * N_;
  _Float16* Shr = Sh + row * N_;
  for (int c = c0; c < c0 + 256; c += 4) {
    float4 v = *(const float4*)&Sr[c];
    v.x *= inv; v.y *= inv; v.z *= inv; v.w *= inv;
    *(float4*)&attn_row[c] = v;          // f32 attn to HBM (the roofline term)
    Pack4h pk;
    pk.h[0] = (_Float16)v.x; pk.h[1] = (_Float16)v.y;
    pk.h[2] = (_Float16)v.z; pk.h[3] = (_Float16)v.w;
    *(unsigned long long*)&Shr[c] = pk.u64;  // f16 P for the PV WMMA
  }
  __syncthreads();

  // ---- out = P @ V : each wave owns a 16-wide DH tile; V tile async-DMA ----
  const int dh0 = wv * 16;
  _Float16* Vsw = Vs + wv * 1024;  // halves: 2 buffers x 512
  const _Float16* Vbase = vt + (size_t)(bh * DH_ + dh0) * N_;
  // first tile: Vs[dh 16][j 32], 16 rows x 4 chunks = 64 chunks / 32 lanes
#pragma unroll
  for (int it = 0; it < 2; ++it) {
    const int i = lane + it * 32;
    const int r = i >> 2, c = (i & 3) * 8;
    async_load_b128(Vsw + r * 32 + c, Vbase + (size_t)r * N_ + c);
  }
  pbuf = 0;
  v8f acco = {};
  for (int j0 = 0; j0 < N_; j0 += 32) {
    const bool more = (j0 + 32) < N_;
    if (more) {
      _Float16* dst = Vsw + (pbuf ^ 1) * 512;
#pragma unroll
      for (int it = 0; it < 2; ++it) {
        const int i = lane + it * 32;
        const int r = i >> 2, c = (i & 3) * 8;
        async_load_b128(dst + r * 32 + c, Vbase + (size_t)r * N_ + j0 + 32 + c);
      }
      wait_async<2>();
    } else {
      wait_async<0>();
    }
    const v16h ap = load_frag_a16(Sh, N_, lane, j0);
    const v16h bv = load_frag_b16(Vsw + pbuf * 512, 32, lane, 0);
    acco = __builtin_amdgcn_wmma_f32_16x16x32_f16(false, ap, false, bv,
                                                  (short)0, acco, false, false);
    lds_wave_fence();
    pbuf ^= 1;
  }

  // ctx layout [B*N, INNER], column h*64 + dh (matches transpose+reshape)
  const int b = bh >> 4, h = bh & 15;
  const int nl = lane & 15, mg = lane >> 4;
#pragma unroll
  for (int r = 0; r < 8; ++r) {
    const int m = i0 + r + 8 * mg;
    ctx[((size_t)b * N_ + m) * INNER_ + h * DH_ + dh0 + nl] = (_Float16)acco[r];
  }
}

// ---------------------------------------------------------------------------
// Host-side launch
// ---------------------------------------------------------------------------
extern "C" void kernel_launch(void* const* d_in, const int* in_sizes, int n_in,
                              void* d_out, int out_size, void* d_ws, size_t ws_size,
                              hipStream_t stream) {
  const float* x    = (const float*)d_in[0];
  const float* psin = (const float*)d_in[1];
  const float* pcos = (const float*)d_in[2];
  const float* Wq   = (const float*)d_in[3];
  const float* Wkv  = (const float*)d_in[4];
  const float* Wo   = (const float*)d_in[5];
  const float* bo   = (const float*)d_in[6];

  float* out_final = (float*)d_out;
  float* attn_out  = (float*)d_out + (size_t)B_ * N_ * D_;

  char* ws = (char*)d_ws;
  _Float16* xh   = (_Float16*)(ws);                 // 8 MB
  _Float16* Wqh  = (_Float16*)(ws + (8u  << 20));   // 2 MB
  _Float16* Wkvh = (_Float16*)(ws + (10u << 20));   // 4 MB
  _Float16* Woh  = (_Float16*)(ws + (14u << 20));   // 2 MB
  float*    qf   = (float*)   (ws + (16u << 20));   // 16 MB
  float*    kvf  = (float*)   (ws + (32u << 20));   // 32 MB
  _Float16* qh   = (_Float16*)(ws + (64u << 20));   // 8 MB
  _Float16* kh   = (_Float16*)(ws + (72u << 20));   // 8 MB
  _Float16* vt   = (_Float16*)(ws + (80u << 20));   // 8 MB [B,H,DH,N]
  _Float16* ctxh = (_Float16*)(ws + (88u << 20));   // 8 MB -> 96 MB total

  const int BN = B_ * N_;

  k_f32_to_f16<<<(BN * D_ / 4) / 256, 256, 0, stream>>>(x, xh, BN * D_ / 4);
  k_f32_to_f16<<<(INNER_ * D_ / 4) / 256, 256, 0, stream>>>(Wq, Wqh, INNER_ * D_ / 4);
  k_f32_to_f16<<<(2 * INNER_ * D_ / 4) / 256, 256, 0, stream>>>(Wkv, Wkvh, 2 * INNER_ * D_ / 4);
  k_f32_to_f16<<<(D_ * INNER_ / 4) / 256, 256, 0, stream>>>(Wo, Woh, D_ * INNER_ / 4);

  k_gemm_wmma<<<dim3(INNER_ / 64, BN / 64), 128, 0, stream>>>(
      xh, Wqh, qf, nullptr, BN, D_, INNER_);
  k_gemm_wmma<<<dim3((2 * INNER_) / 64, BN / 64), 128, 0, stream>>>(
      xh, Wkvh, kvf, nullptr, BN, D_, 2 * INNER_);

  {
    const int total = B_ * N_ * H_ * (DH_ / 2);
    k_rope_pack<<<(total + 255) / 256, 256, 0, stream>>>(qf, kvf, psin, pcos,
                                                         qh, kh, vt);
  }

  const int smem_bytes = 224256;  // 219 KB of the 320 KB WGP LDS
  static_cast<void>(hipFuncSetAttribute(
      (const void*)k_attention, hipFuncAttributeMaxDynamicSharedMemorySize,
      smem_bytes));
  k_attention<<<B_ * H_ * (N_ / 16), 128, smem_bytes, stream>>>(
      qh, kh, vt, attn_out, ctxh);

  k_gemm_wmma<<<dim3(D_ / 64, BN / 64), 128, 0, stream>>>(
      ctxh, Woh, out_final, bo, BN, INNER_, D_);
}
